// BlockSparseAttention_59588376264815
// MI455X (gfx1250) — matrix operand order repaired and checked by reading, not compiled, not run on hardware
//
#include <hip/hip_runtime.h>

typedef __bf16 bf16_t;
typedef __attribute__((ext_vector_type(16))) __bf16 v16bf;
typedef __attribute__((ext_vector_type(8)))  float  v8f;
typedef __attribute__((ext_vector_type(4)))  int    v4i;

#define B_  2
#define S_  2048
#define E_  1024
#define H_  16
#define D_  64
#define NB_ 32
#define M_  (B_ * S_)   // 4096 rows in the flattened GEMMs

// ---------------------------------------------------------------------------
// CDNA5 async global->LDS copy (ASYNCcnt path), guarded so the kernel still
// compiles if this toolchain does not expose the builtins.
// Probe round 2: builtin exists; param0 = AS1 (global) v4i*, param1 = AS3 v4i*.
// ---------------------------------------------------------------------------
#if __has_builtin(__builtin_amdgcn_global_load_async_to_lds_b128) && \
    __has_builtin(__builtin_amdgcn_s_wait_asynccnt)
#define ASYNC_LDS 1
typedef __attribute__((address_space(1))) v4i gas_v4i;
typedef __attribute__((address_space(3))) v4i las_v4i;
#else
#define ASYNC_LDS 0
#endif

__device__ __forceinline__ void cp16_to_lds(const bf16_t* g, bf16_t* l) {
#if ASYNC_LDS
  __builtin_amdgcn_global_load_async_to_lds_b128(
      (gas_v4i*)g, (las_v4i*)l, /*offset=*/0, /*cpol=*/0);
#else
  *(uint4*)l = *(const uint4*)g;
#endif
}
__device__ __forceinline__ void cp_wait() {
#if ASYNC_LDS
  __builtin_amdgcn_s_wait_asynccnt(0);
#endif
}

// ---------------------------------------------------------------------------
// WMMA helpers (CDNA5: V_WMMA_F32_16X16X32_BF16, f32 accumulate)
// ---------------------------------------------------------------------------
__device__ __forceinline__ v8f wmma_bf16(v16bf a, v16bf b, v8f c) {
  return __builtin_amdgcn_wmma_f32_16x16x32_bf16(
      /*neg_a=*/false, a, /*neg_b=*/false, b,
      /*c_mod=*/(short)0, c, /*reuse_a=*/false, /*reuse_b=*/false);
}

// A fragment (16x32 bf16) from an LDS tile stored row-major [64][64].
// ISA layout: lane half h, m = lane&15; elems 0..7 = K(k0 + 8h + 0..7),
// elems 8..15 = K(k0 + 16 + 8h + 0..7).
__device__ __forceinline__ v16bf lds_load_a(const bf16_t* t, int m0, int k0) {
  const int lane = threadIdx.x & 31;
  const bf16_t* p = t + (m0 + (lane & 15)) * 64 + k0 + ((lane >> 4) << 3);
  v16bf a;
#pragma unroll
  for (int i = 0; i < 8; ++i) { a[i] = p[i]; a[i + 8] = p[16 + i]; }
  return a;
}

// B fragment (32x16 bf16). LDS holds B^T row-major: [n][k], ld = 64.
// ISA layout: lane half h; elems 0..7 = K(k0 + 8h + 0..7), elems 8..15 = K(k0 + 16 + 8h + 0..7).
__device__ __forceinline__ v16bf lds_load_b(const bf16_t* t, int n0, int k0) {
  const int lane = threadIdx.x & 31;
  const bf16_t* p = t + (n0 + (lane & 15)) * 64 + k0 + ((lane >> 4) << 3);
  v16bf b;
#pragma unroll
  for (int i = 0; i < 8; ++i) { b[i] = p[i]; b[i + 8] = p[16 + i]; }
  return b;
}

// ---------------------------------------------------------------------------
// Conversions
// ---------------------------------------------------------------------------
__global__ void cvt_x_kernel(const float* __restrict__ in, bf16_t* __restrict__ out) {
  const int i = (blockIdx.x * 256 + threadIdx.x) * 4;  // n is multiple of 4
  const float4 v = *(const float4*)(in + i);
  out[i + 0] = (bf16_t)v.x;  out[i + 1] = (bf16_t)v.y;
  out[i + 2] = (bf16_t)v.z;  out[i + 3] = (bf16_t)v.w;
}

// Wt[n][k] = (bf16)W[k][n]   (E_ x E_), tiled via LDS for coalescing.
__global__ void transpose_cvt_kernel(const float* __restrict__ W, bf16_t* __restrict__ Wt) {
  __shared__ float tile[32][33];
  const int bx = blockIdx.x * 32, by = blockIdx.y * 32;
  const int tx = threadIdx.x, ty = threadIdx.y;          // block (32, 8)
#pragma unroll
  for (int i = 0; i < 32; i += 8)
    tile[ty + i][tx] = W[(size_t)(by + ty + i) * E_ + bx + tx];
  __syncthreads();
#pragma unroll
  for (int i = 0; i < 32; i += 8)
    Wt[(size_t)(bx + ty + i) * E_ + by + tx] = (bf16_t)tile[tx][ty + i];
}

// ---------------------------------------------------------------------------
// Shared GEMM core: C(64x64 per WG) = A(row-major) @ Bt^T with double-buffered
// LDS tiles and async global->LDS copies.  4 waves, each owns a 16-row strip.
// ---------------------------------------------------------------------------
__device__ __forceinline__ void gemm_core(const bf16_t* __restrict__ Arow,
                                          const bf16_t* __restrict__ Bt,
                                          int m0, int n0, int mrow,
                                          bf16_t (*As)[4096], bf16_t (*Bs)[4096],
                                          v8f acc[4]) {
  const int tid = threadIdx.x;
  // prologue: tile 0
#pragma unroll
  for (int i = tid; i < 512; i += 128) {
    const int r = i >> 3, c = (i & 7) << 3;
    cp16_to_lds(Arow + (size_t)(m0 + r) * E_ + c, &As[0][r * 64 + c]);
    cp16_to_lds(Bt + (size_t)(n0 + r) * E_ + c, &Bs[0][r * 64 + c]);
  }
  cp_wait();
  __syncthreads();

  for (int t = 0; t < E_ / 64; ++t) {
    const int cur = t & 1;
    if (t + 1 < E_ / 64) {                      // prefetch next tile (async)
      const int kt = (t + 1) * 64;
#pragma unroll
      for (int i = tid; i < 512; i += 128) {
        const int r = i >> 3, c = (i & 7) << 3;
        cp16_to_lds(Arow + (size_t)(m0 + r) * E_ + kt + c, &As[cur ^ 1][r * 64 + c]);
        cp16_to_lds(Bt + (size_t)(n0 + r) * E_ + kt + c, &Bs[cur ^ 1][r * 64 + c]);
      }
    }
#pragma unroll
    for (int kk = 0; kk < 64; kk += 32) {
      const v16bf a = lds_load_a(As[cur], mrow, kk);
#pragma unroll
      for (int nt = 0; nt < 4; ++nt)
        acc[nt] = wmma_bf16(a, lds_load_b(Bs[cur], nt * 16, kk), acc[nt]);
    }
    cp_wait();                                  // next buffer resident
    __syncthreads();                            // all waves done reading cur
  }
}

// ---------------------------------------------------------------------------
// Projection GEMM: y = x @ W + b  -> bf16 [B,H,S,D], scaled (q: 1/sqrt(D)).
// grid (M_/64, E_/64), block 128.
// ---------------------------------------------------------------------------
__global__ __launch_bounds__(128) void proj_kernel(
    const bf16_t* __restrict__ xb,   // [M_][E_]
    const bf16_t* __restrict__ Wt,   // [E_][E_]  (W transposed)
    const float*  __restrict__ bias, // [E_]
    bf16_t* __restrict__ out,        // [B,H,S,D]
    float scale) {
  __shared__ __align__(16) bf16_t As[2][4096];
  __shared__ __align__(16) bf16_t Bs[2][4096];
  const int m0 = blockIdx.x * 64, n0 = blockIdx.y * 64;
  const int lane = threadIdx.x & 31, half = lane >> 4, l16 = lane & 15;
  const int mrow = (threadIdx.x >> 5) * 16;

  v8f acc[4] = {};
  gemm_core(xb, Wt, m0, n0, mrow, As, Bs, acc);

#pragma unroll
  for (int nt = 0; nt < 4; ++nt) {
    const int n = n0 + nt * 16 + l16;
    const int h = n >> 6, d = n & 63;
    const float bv = bias[n];
#pragma unroll
    for (int j = 0; j < 8; ++j) {
      const int m = m0 + mrow + half * 8 + j;
      const int b = m >> 11, s = m & (S_ - 1);
      out[(((size_t)(b * H_ + h)) * S_ + s) * D_ + d] = (bf16_t)((acc[nt][j] + bv) * scale);
    }
  }
}

// ---------------------------------------------------------------------------
// Output GEMM: d_out = accb @ Wo + bo  (f32 output, row-major [M_][E_])
// ---------------------------------------------------------------------------
__global__ __launch_bounds__(128) void outproj_kernel(
    const bf16_t* __restrict__ ab,   // [M_][E_] bf16
    const bf16_t* __restrict__ Wt,   // [E_][E_] (Wo transposed)
    const float*  __restrict__ bias,
    float* __restrict__ out) {
  __shared__ __align__(16) bf16_t As[2][4096];
  __shared__ __align__(16) bf16_t Bs[2][4096];
  const int m0 = blockIdx.x * 64, n0 = blockIdx.y * 64;
  const int lane = threadIdx.x & 31, half = lane >> 4, l16 = lane & 15;
  const int mrow = (threadIdx.x >> 5) * 16;

  v8f acc[4] = {};
  gemm_core(ab, Wt, m0, n0, mrow, As, Bs, acc);

#pragma unroll
  for (int nt = 0; nt < 4; ++nt) {
    const int n = n0 + nt * 16 + l16;
    const float bv = bias[n];
#pragma unroll
    for (int j = 0; j < 8; ++j) {
      const int m = m0 + mrow + half * 8 + j;
      out[(size_t)m * E_ + n] = acc[nt][j] + bv;
    }
  }
}

// ---------------------------------------------------------------------------
// Block-sparse attention. grid (NB_, B_*H_), block 128 (4 waves).
// Each workgroup owns query block-row bi of one (b,h); scans the active-block
// list in fixed order (deterministic segment sum, no float atomics), does
// block-local softmax per the reference, accumulates O in registers.
// ---------------------------------------------------------------------------
__global__ __launch_bounds__(128) void attn_kernel(
    const bf16_t* __restrict__ qb,   // [B,H,S,D] (already scaled)
    const bf16_t* __restrict__ kb,
    const bf16_t* __restrict__ vb,
    const int* __restrict__ brows, const int* __restrict__ bcols, int nact,
    bf16_t* __restrict__ accb) {     // [B,S,E]
  __shared__ __align__(16) bf16_t q_s[64 * 64];
  __shared__ __align__(16) bf16_t k_s[64 * 64];
  __shared__ __align__(16) bf16_t vT_s[64 * 64];
  __shared__ __align__(16) bf16_t p_s[64 * 64];

  const int bi = blockIdx.x;                 // query block-row
  const int bh = blockIdx.y;                 // b*H_ + h
  const int b = bh >> 4, h = bh & (H_ - 1);
  const size_t base_bh = (size_t)bh * S_ * D_;
  const int tid = threadIdx.x;
  const int wave = tid >> 5, lane = tid & 31, half = lane >> 4, l16 = lane & 15;
  const int mrow = wave * 16;

  {  // q block: contiguous 64*64 bf16 (async copy; waited before first use)
    const bf16_t* src = qb + base_bh + (size_t)bi * 4096;
#pragma unroll
    for (int i = tid; i < 512; i += 128) cp16_to_lds(src + i * 8, &q_s[i * 8]);
  }

  v8f o[4] = {};
  for (int a = 0; a < nact; ++a) {
    if (brows[a] != bi) continue;            // uniform across workgroup
    const int bj = bcols[a];
    __syncthreads();                         // prev-iter LDS reads done
    {
      const bf16_t* ks = kb + base_bh + (size_t)bj * 4096;
#pragma unroll
      for (int i = tid; i < 512; i += 128) cp16_to_lds(ks + i * 8, &k_s[i * 8]);
      const bf16_t* vs = vb + base_bh + (size_t)bj * 4096;
      for (int i = tid; i < 4096; i += 128) {      // v^T staging: [d][key]
        const int r = i >> 6, d = i & 63;
        vT_s[d * 64 + r] = vs[i];
      }
    }
    cp_wait();
    __syncthreads();

    // scores = q @ k^T  (K = D = 64)
    v8f sc[4] = {};
#pragma unroll
    for (int kk = 0; kk < 64; kk += 32) {
      const v16bf aq = lds_load_a(q_s, mrow, kk);
#pragma unroll
      for (int nt = 0; nt < 4; ++nt)
        sc[nt] = wmma_bf16(aq, lds_load_b(k_s, nt * 16, kk), sc[nt]);
    }

    // block-local softmax over 64 keys per row; rows split 0..7 / 8..15 by half
#pragma unroll
    for (int j = 0; j < 8; ++j) {
      float mx = fmaxf(fmaxf(sc[0][j], sc[1][j]), fmaxf(sc[2][j], sc[3][j]));
#pragma unroll
      for (int msk = 8; msk; msk >>= 1) mx = fmaxf(mx, __shfl_xor(mx, msk, 16));
      float pv[4], sm = 0.0f;
#pragma unroll
      for (int nt = 0; nt < 4; ++nt) { pv[nt] = __expf(sc[nt][j] - mx); sm += pv[nt]; }
#pragma unroll
      for (int msk = 8; msk; msk >>= 1) sm += __shfl_xor(sm, msk, 16);
      const float inv = 1.0f / sm;
      const int mg = mrow + half * 8 + j;
#pragma unroll
      for (int nt = 0; nt < 4; ++nt)
        p_s[mg * 64 + nt * 16 + l16] = (bf16_t)(pv[nt] * inv);
    }
    __syncthreads();

    // O += P @ V   (B fragments from v^T)
#pragma unroll
    for (int kk = 0; kk < 64; kk += 32) {
      const v16bf ap = lds_load_a(p_s, mrow, kk);
#pragma unroll
      for (int dt = 0; dt < 4; ++dt)
        o[dt] = wmma_bf16(ap, lds_load_b(vT_s, dt * 16, kk), o[dt]);
    }
  }

  // write [B,S,E] bf16 (every (bi,bh) covers its rows; empty rows write zeros)
#pragma unroll
  for (int dt = 0; dt < 4; ++dt) {
    const int dcol = dt * 16 + l16;
#pragma unroll
    for (int j = 0; j < 8; ++j) {
      const int s = bi * 64 + mrow + half * 8 + j;
      accb[((size_t)b * S_ + s) * E_ + h * D_ + dcol] = (bf16_t)o[dt][j];
    }
  }
}

// ---------------------------------------------------------------------------
extern "C" void kernel_launch(void* const* d_in, const int* in_sizes, int n_in,
                              void* d_out, int out_size, void* d_ws, size_t ws_size,
                              hipStream_t stream) {
  (void)n_in; (void)out_size; (void)ws_size;
  const float* x  = (const float*)d_in[0];
  const float* Wq = (const float*)d_in[1];
  const float* bq = (const float*)d_in[2];
  const float* Wk = (const float*)d_in[3];
  const float* bk = (const float*)d_in[4];
  const float* Wv = (const float*)d_in[5];
  const float* bv = (const float*)d_in[6];
  const float* Wo = (const float*)d_in[7];
  const float* bo = (const float*)d_in[8];
  const int* brows = (const int*)d_in[9];
  const int* bcols = (const int*)d_in[10];
  const int nact = in_sizes[9];
  float* out = (float*)d_out;

  char* p = (char*)d_ws;
  auto carve = [&](size_t bytes) {
    char* r = p; p += (bytes + 255) & ~(size_t)255; return r;
  };
  bf16_t* xb  = (bf16_t*)carve((size_t)M_ * E_ * 2);
  bf16_t* Wqt = (bf16_t*)carve((size_t)E_ * E_ * 2);
  bf16_t* Wkt = (bf16_t*)carve((size_t)E_ * E_ * 2);
  bf16_t* Wvt = (bf16_t*)carve((size_t)E_ * E_ * 2);
  bf16_t* Wot = (bf16_t*)carve((size_t)E_ * E_ * 2);
  bf16_t* qbf = (bf16_t*)carve((size_t)B_ * H_ * S_ * D_ * 2);
  bf16_t* kbf = (bf16_t*)carve((size_t)B_ * H_ * S_ * D_ * 2);
  bf16_t* vbf = (bf16_t*)carve((size_t)B_ * H_ * S_ * D_ * 2);
  bf16_t* acb = (bf16_t*)carve((size_t)B_ * S_ * E_ * 2);

  // 1) precision conversion / weight transpose
  cvt_x_kernel<<<(M_ * E_) / 1024, 256, 0, stream>>>(x, xb);
  dim3 tgrid(E_ / 32, E_ / 32), tblk(32, 8);
  transpose_cvt_kernel<<<tgrid, tblk, 0, stream>>>(Wq, Wqt);
  transpose_cvt_kernel<<<tgrid, tblk, 0, stream>>>(Wk, Wkt);
  transpose_cvt_kernel<<<tgrid, tblk, 0, stream>>>(Wv, Wvt);
  transpose_cvt_kernel<<<tgrid, tblk, 0, stream>>>(Wo, Wot);

  // 2) q/k/v projections (q pre-scaled by 1/sqrt(D) = 0.125)
  dim3 pgrid(M_ / 64, E_ / 64);
  proj_kernel<<<pgrid, 128, 0, stream>>>(xb, Wqt, bq, qbf, 0.125f);
  proj_kernel<<<pgrid, 128, 0, stream>>>(xb, Wkt, bk, kbf, 1.0f);
  proj_kernel<<<pgrid, 128, 0, stream>>>(xb, Wvt, bv, vbf, 1.0f);

  // 3) block-sparse attention with per-block softmax + deterministic segment sum
  dim3 agrid(NB_, B_ * H_);
  attn_kernel<<<agrid, 128, 0, stream>>>(qbf, kbf, vbf, brows, bcols, nact, acb);

  // 4) output projection -> f32 d_out
  outproj_kernel<<<pgrid, 128, 0, stream>>>(acb, Wot, bo, out);
}
